// CausalPerformer_4320737100672
// MI455X (gfx1250) — matrix operand order repaired and checked by reading, not compiled, not run on hardware
//
#include <hip/hip_runtime.h>
#include <hip/hip_fp16.h>

typedef _Float16 f16;
typedef __attribute__((ext_vector_type(16))) _Float16 v16h;
typedef __attribute__((ext_vector_type(8)))  float    v8f;
typedef __attribute__((ext_vector_type(4)))  int      i4;

// ---- model dims (from reference) ----
constexpr int Bz = 4, Ez = 1024, Cz = 4, Vz = 128;
constexpr int EMBz = 64, PEz = 64, Dz = 512, Hz = 8, HLz = 4, DHz = 64;
constexpr int Mz = 256, DFFz = 2048, Lz = 6;
constexpr int Sz = Ez * Cz;          // 4096
constexpr long BSz = (long)Bz * Sz;  // 16384
constexpr int CHz = 256, Wz = 256;   // performer chunk, local window
constexpr int BGz = Bz * (Hz - HLz); // 16 performer (b,h) batches
constexpr int NCH = Sz / CHz;        // 16 chunks

struct __align__(16) h8 { f16 v[8]; };

#define AS1 __attribute__((address_space(1)))
#define AS3 __attribute__((address_space(3)))

#if defined(__gfx1250__) && __has_builtin(__builtin_amdgcn_global_load_async_to_lds_b128)
#define HAVE_ASYNC 1
#else
#define HAVE_ASYNC 0
#endif

// 16-byte global -> LDS copy; async (ASYNCcnt-tracked) when available.
static __device__ __forceinline__ void cp16_g2l(const f16* g, f16* l)
{
#if HAVE_ASYNC
    __builtin_amdgcn_global_load_async_to_lds_b128(
        (AS1 i4*)(void*)g, (AS3 i4*)(void*)l, 0, 0);
#else
    *(h8*)l = *(const h8*)g;
#endif
}

static __device__ __forceinline__ void async_wait()
{
#if HAVE_ASYNC
#if __has_builtin(__builtin_amdgcn_s_wait_asynccnt)
    __builtin_amdgcn_s_wait_asynccnt(0);
#else
    asm volatile("s_wait_asynccnt 0" ::: "memory");
#endif
#endif
}

// ============================================================
// Generic batched WMMA GEMM:  Out[b] = A[b] (MxK) * B[b] (KxN) (+bias)
// A,B are f16.  TA: A stored [K][M].  TB: B stored [N][K].
// O16F: store f16 output; ACCF: accumulate into f32 output.
// Block tile 128x64x32; 8 waves as 4(M) x 2(N); 32x32 per wave
// (2x2 WMMA tiles with A/B fragment reuse -> 4 v_wmma per K-step).
// ============================================================
template<bool TA, bool TB, bool O16F, bool ACCF>
__global__ __launch_bounds__(256)
void gemm_wmma_kernel(const f16* __restrict__ Ag, long lda, long sA,
                      const f16* __restrict__ Bg, long ldb, long sB,
                      const float* __restrict__ bias,
                      void* __restrict__ Outv, long ldc, long sC, int K)
{
    __shared__ __align__(16) f16 Al[128][32];
    __shared__ __align__(16) f16 Bl[64][32];   // [n][k]
    const int tid  = threadIdx.x;
    const int lane = tid & 31;
    const int w    = tid >> 5;
    const int wm   = w & 3, wn = w >> 2;
    const long m0  = (long)blockIdx.x * 128;
    const long n0  = (long)blockIdx.y * 64;
    const long bz  = blockIdx.z;
    Ag += bz * sA;  Bg += bz * sB;
    float* O32 = (float*)Outv + bz * sC;
    f16*   O16 = (f16*)Outv + bz * sC;

    const int r  = lane & 15;
    const int hi = lane >> 4;
    v8f acc00 = {}, acc01 = {}, acc10 = {}, acc11 = {};

    for (int k0 = 0; k0 < K; k0 += 32) {
        // ---- stage A tile -> Al[m][k] (128x32) ----
        if (!TA) {
            int m  = tid >> 1;
            int ks = (tid & 1) * 16;
            const f16* src = Ag + (m0 + m) * lda + k0 + ks;
            cp16_g2l(src,     &Al[m][ks]);
            cp16_g2l(src + 8, &Al[m][ks + 8]);
        } else {
            int k  = tid & 31;
            int ms = (tid >> 5) * 16;
            const f16* src = Ag + (long)(k0 + k) * lda + m0 + ms;
            h8 t0 = *(const h8*)src;
            h8 t1 = *(const h8*)(src + 8);
            #pragma unroll
            for (int j = 0; j < 8; ++j) {
                Al[ms + j][k]     = t0.v[j];
                Al[ms + 8 + j][k] = t1.v[j];
            }
        }
        // ---- stage B tile -> Bl[n][k] (64x32) ----
        if (!TB) {
            int k  = tid & 31;
            int ng = (tid >> 5) * 8;
            const f16* src = Bg + (long)(k0 + k) * ldb + n0 + ng;
            h8 t = *(const h8*)src;
            #pragma unroll
            for (int j = 0; j < 8; ++j) Bl[ng + j][k] = t.v[j];
        } else {
            int n  = tid >> 2;
            int ks = (tid & 3) * 8;
            cp16_g2l(Bg + (long)(n0 + n) * ldb + k0 + ks, &Bl[n][ks]);
        }
        // prefetch next tiles (gfx1250 global_prefetch_b8 path)
        if (k0 + 32 < K) {
            __builtin_prefetch(Ag + (TA ? ((long)(k0 + 32 + (tid & 31)) * lda + m0)
                                        : ((m0 + (tid >> 1)) * lda + k0 + 32)), 0, 3);
            __builtin_prefetch(Bg + (TB ? ((long)(n0 + (tid >> 2)) * ldb + k0 + 32)
                                        : ((long)(k0 + 32 + (tid & 31)) * ldb + n0)), 0, 3);
        }
        async_wait();
        __syncthreads();
        // ---- fragments (contiguous k = 16*hi + i, same order in A and B) ----
        v16h a0, a1, b0, b1;
        {
            const f16* ap0 = &Al[wm * 32 + r][hi * 16];
            const f16* ap1 = &Al[wm * 32 + 16 + r][hi * 16];
            #pragma unroll
            for (int i = 0; i < 8; ++i) {
                a0[i] = ap0[i]; a0[8 + i] = ap0[8 + i];
                a1[i] = ap1[i]; a1[8 + i] = ap1[8 + i];
            }
            const f16* bp0 = &Bl[wn * 32 + r][hi * 16];
            const f16* bp1 = &Bl[wn * 32 + 16 + r][hi * 16];
            #pragma unroll
            for (int i = 0; i < 16; ++i) { b0[i] = bp0[i]; b1[i] = bp1[i]; }
        }
        acc00 = __builtin_amdgcn_wmma_f32_16x16x32_f16(false, a0, false, b0, (short)0, acc00, false, false);
        acc01 = __builtin_amdgcn_wmma_f32_16x16x32_f16(false, a0, false, b1, (short)0, acc01, false, false);
        acc10 = __builtin_amdgcn_wmma_f32_16x16x32_f16(false, a1, false, b0, (short)0, acc10, false, false);
        acc11 = __builtin_amdgcn_wmma_f32_16x16x32_f16(false, a1, false, b1, (short)0, acc11, false, false);
        __syncthreads();
    }
    // ---- epilogue: 2x2 tiles ----
    #pragma unroll
    for (int mi = 0; mi < 2; ++mi) {
        #pragma unroll
        for (int ni = 0; ni < 2; ++ni) {
            const v8f& a = mi ? (ni ? acc11 : acc10) : (ni ? acc01 : acc00);
            long col = n0 + wn * 32 + ni * 16 + r;
            float bb = bias ? bias[col] : 0.f;
            #pragma unroll
            for (int i = 0; i < 8; ++i) {
                long row = m0 + wm * 32 + mi * 16 + hi * 8 + i;
                float v  = a[i] + bb;
                long idx = row * ldc + col;
                if (O16F)      O16[idx] = (f16)v;
                else if (ACCF) O32[idx] += v;
                else           O32[idx] = v;
            }
        }
    }
}

// ============================================================
// f32 -> f16 conversion (weights)
// ============================================================
__global__ void cvt_kernel(const float* __restrict__ s, f16* __restrict__ d, long n)
{
    long i = (long)blockIdx.x * 256 + threadIdx.x;
    long st = (long)gridDim.x * 256;
    for (; i < n; i += st) d[i] = (f16)s[i];
}

// ============================================================
// Embedding gather + PE concat -> A0 (BS x 128) f16
// ============================================================
__global__ void embed_kernel(const int* __restrict__ tgt, const float* __restrict__ emb,
                             const float* __restrict__ pe, f16* __restrict__ A0)
{
    long t = (long)blockIdx.x * 256 + threadIdx.x;   // BS*128
    long row = t >> 7; int c = (int)(t & 127);
    long s = row % Sz;
    float v;
    if (c < EMBz) {
        int cpos = (int)(s & 3);
        int id = tgt[row];
        v = emb[((long)cpos * Vz + id) * EMBz + c];
    } else {
        v = pe[s * PEz + (c - EMBz)];
    }
    A0[t] = (f16)v;
}

// shift-by-one, add sos and sinusoid PE; init x1 = x2 = x
__global__ void assemble_kernel(const float* __restrict__ lin, const float* __restrict__ sos,
                                float* __restrict__ x1, float* __restrict__ x2)
{
    long t = (long)blockIdx.x * 256 + threadIdx.x;   // BS*512
    long row = t >> 9; int d = (int)(t & 511);
    long s = row % Sz;
    float base = (s == 0) ? sos[d] : lin[(row - 1) * Dz + d];
    float pv;
    if (d < 256) pv = __sinf((float)s * __powf(10000.f, -(float)d / 256.f));
    else         pv = __cosf((float)s * __powf(10000.f, -(float)(d - 256) / 256.f));
    float x = base + pv;
    x1[t] = x; x2[t] = x;
}

// ============================================================
// LayerNorm over D=512, one wave per row, f16 output
// ============================================================
__global__ __launch_bounds__(256)
void ln_kernel(const float* __restrict__ x, const float* __restrict__ g,
               const float* __restrict__ b, f16* __restrict__ out)
{
    int lane = threadIdx.x & 31;
    long row = (long)blockIdx.x * 8 + (threadIdx.x >> 5);
    const float* xr = x + row * Dz;
    float vals[16], s = 0.f;
    #pragma unroll
    for (int i = 0; i < 16; ++i) { vals[i] = xr[lane + i * 32]; s += vals[i]; }
    #pragma unroll
    for (int o = 16; o; o >>= 1) s += __shfl_xor(s, o, 32);
    float mu = s * (1.f / Dz), vv = 0.f;
    #pragma unroll
    for (int i = 0; i < 16; ++i) { float d = vals[i] - mu; vv += d * d; }
    #pragma unroll
    for (int o = 16; o; o >>= 1) vv += __shfl_xor(vv, o, 32);
    float rstd = rsqrtf(vv * (1.f / Dz) + 1e-5f);
    #pragma unroll
    for (int i = 0; i < 16; ++i) {
        int d = lane + i * 32;
        out[row * Dz + d] = (f16)((vals[i] - mu) * rstd * g[d] + b[d]);
    }
}

// final LN on y = 0.5*(x1+x2), f32 output
__global__ __launch_bounds__(256)
void final_ln_kernel(const float* __restrict__ xa, const float* __restrict__ xb,
                     const float* __restrict__ g, const float* __restrict__ b,
                     float* __restrict__ out)
{
    int lane = threadIdx.x & 31;
    long row = (long)blockIdx.x * 8 + (threadIdx.x >> 5);
    float vals[16], s = 0.f;
    #pragma unroll
    for (int i = 0; i < 16; ++i) {
        long idx = row * Dz + lane + i * 32;
        vals[i] = 0.5f * (xa[idx] + xb[idx]); s += vals[i];
    }
    #pragma unroll
    for (int o = 16; o; o >>= 1) s += __shfl_xor(s, o, 32);
    float mu = s * (1.f / Dz), vv = 0.f;
    #pragma unroll
    for (int i = 0; i < 16; ++i) { float d = vals[i] - mu; vv += d * d; }
    #pragma unroll
    for (int o = 16; o; o >>= 1) vv += __shfl_xor(vv, o, 32);
    float rstd = rsqrtf(vv * (1.f / Dz) + 1e-5f);
    #pragma unroll
    for (int i = 0; i < 16; ++i) {
        int d = lane + i * 32;
        out[row * Dz + d] = (vals[i] - mu) * rstd * g[d] + b[d];
    }
}

// ============================================================
// QKV split + rotary; local heads -> qh/kh/vh [bh16][S][64];
// performer heads (pre-scaled by 64^-0.25) -> qg/kg/vg [bg16][S][64];
// per-row diag = 0.5*sum(xn^2).
// ============================================================
__global__ __launch_bounds__(512)
void rotary_split_kernel(const f16* __restrict__ qkv,
                         f16* __restrict__ qh, f16* __restrict__ kh, f16* __restrict__ vh,
                         f16* __restrict__ qg, f16* __restrict__ kg, f16* __restrict__ vg,
                         float* __restrict__ diagq, float* __restrict__ diagk)
{
    __shared__ float sdq[4], sdk[4];
    long row = blockIdx.x;                 // b*S + s
    long b = row / Sz, s = row % Sz;
    int t = threadIdx.x;
    int h = t >> 6, d = t & 63;
    if (t < 4) { sdq[t] = 0.f; sdk[t] = 0.f; }
    __syncthreads();
    const f16* base = qkv + row * (3 * Dz);
    int pi = d >> 1;
    float ang = (float)s * __powf(10000.f, -(float)pi / 32.f);
    float sn = __sinf(ang), cs = __cosf(ang);
    int de = d & ~1;
    float q0 = (float)base[h * 64 + de],        q1 = (float)base[h * 64 + de + 1];
    float k0 = (float)base[Dz + h * 64 + de],   k1 = (float)base[Dz + h * 64 + de + 1];
    float vv = (float)base[2 * Dz + h * 64 + d];
    float qr = (d & 1) ? (q1 * cs + q0 * sn) : (q0 * cs - q1 * sn);
    float kr = (d & 1) ? (k1 * cs + k0 * sn) : (k0 * cs - k1 * sn);
    if (h < HLz) {
        long o = (((long)b * HLz + h) * Sz + s) * 64 + d;
        qh[o] = (f16)qr; kh[o] = (f16)kr; vh[o] = (f16)vv;
    } else {
        int hh = h - HLz;
        long bg = b * 4 + hh;
        long o = (bg * Sz + s) * 64 + d;
        float qs = qr * 0.35355339f, ks = kr * 0.35355339f;   // 64^-0.25
        qg[o] = (f16)qs; kg[o] = (f16)ks; vg[o] = (f16)vv;
        atomicAdd(&sdq[hh], qs * qs);
        atomicAdd(&sdk[hh], ks * ks);
    }
    __syncthreads();
    if (t < 4) {
        long bg = b * 4 + t;
        diagq[bg * Sz + s] = 0.5f * sdq[t];
        diagk[bg * Sz + s] = 0.5f * sdk[t];
    }
}

// ============================================================
// Local windowed attention (heads 0..3), WMMA scores + softmax.
// grid: (qtile 16, window 16, b*HL 16), 256 threads.
// ============================================================
__global__ __launch_bounds__(256)
void local_attn_kernel(const f16* __restrict__ qh, const f16* __restrict__ kh,
                       const f16* __restrict__ vh, f16* __restrict__ outA)
{
    __shared__ float Sc[16][512];
    __shared__ f16   P[16][512];
    int qt = blockIdx.x, n = blockIdx.y, bh = blockIdx.z;
    int b = bh >> 2, h = bh & 3;
    int lane = threadIdx.x & 31, w = threadIdx.x >> 5;
    int r = lane & 15, hi = lane >> 4;
    const f16* qbase = qh + (long)bh * Sz * 64;
    const f16* kbase = kh + (long)bh * Sz * 64;
    const f16* vbase = vh + (long)bh * Sz * 64;

    // phase 1: score tiles 16x16 via WMMA (K=64 in two steps)
    for (int kt = w; kt < 32; kt += 8) {
        int j0 = kt * 16;
        bool dead = (n == 0 && j0 < Wz);
        v8f acc = {};
        if (!dead) {
            int jc = j0 + r;
            long skey = (jc < Wz) ? ((long)(n - 1) * Wz + jc) : ((long)n * Wz + jc - Wz);
            long sq = (long)n * Wz + qt * 16 + r;
            #pragma unroll
            for (int ks = 0; ks < 2; ++ks) {
                v16h av, bv;
                const f16* ap = qbase + sq * 64 + ks * 32 + hi * 16;
                #pragma unroll
                for (int i = 0; i < 8; ++i) { av[i] = ap[i]; av[8 + i] = ap[8 + i]; }
                const f16* bp = kbase + skey * 64 + ks * 32 + hi * 16;
                #pragma unroll
                for (int i = 0; i < 16; ++i) bv[i] = bp[i];
                acc = __builtin_amdgcn_wmma_f32_16x16x32_f16(false, av, false, bv,
                                                             (short)0, acc, false, false);
            }
        }
        #pragma unroll
        for (int i = 0; i < 8; ++i) {
            int m = hi * 8 + i;
            int j = j0 + r;
            int iq = qt * 16 + m;
            float v = dead ? -1e9f : acc[i] * 0.125f;   // dh^-0.5
            if (j > iq + Wz) v = -1e9f;
            Sc[m][j] = v;
        }
    }
    __syncthreads();
    // phase 2: softmax per row (wave handles 2 rows)
    for (int rr = 0; rr < 2; ++rr) {
        int m = w * 2 + rr;
        float vals[16], mx = -1e30f;
        #pragma unroll
        for (int i = 0; i < 16; ++i) { vals[i] = Sc[m][lane + i * 32]; mx = fmaxf(mx, vals[i]); }
        #pragma unroll
        for (int o = 16; o; o >>= 1) mx = fmaxf(mx, __shfl_xor(mx, o, 32));
        float ssum = 0.f;
        #pragma unroll
        for (int i = 0; i < 16; ++i) { vals[i] = __expf(vals[i] - mx); ssum += vals[i]; }
        #pragma unroll
        for (int o = 16; o; o >>= 1) ssum += __shfl_xor(ssum, o, 32);
        float inv = 1.f / ssum;
        #pragma unroll
        for (int i = 0; i < 16; ++i) P[m][lane + i * 32] = (f16)(vals[i] * inv);
    }
    __syncthreads();
    // phase 3: out = P x V (scalar; tiny fraction of total flops)
    for (int e = threadIdx.x; e < 16 * 64; e += 256) {
        int m = e >> 6, d = e & 63;
        float accv = 0.f;
        for (int j = 0; j < 2 * Wz; ++j) {
            if (n == 0 && j < Wz) continue;
            long skey = (j < Wz) ? ((long)(n - 1) * Wz + j) : ((long)n * Wz + j - Wz);
            accv += (float)P[m][j] * (float)vbase[skey * 64 + d];
        }
        long sq = (long)n * Wz + qt * 16 + m;
        outA[(((long)b * Sz + sq) * Hz + h) * DHz + d] = (f16)accv;
    }
}

// ============================================================
// Performer feature maps
// ============================================================
__global__ __launch_bounds__(256)
void qfeat_kernel(const float* __restrict__ xd, const float* __restrict__ diag,
                  f16* __restrict__ qp)
{
    long grow = (long)blockIdx.x * 8 + (threadIdx.x >> 5);
    int lane = threadIdx.x & 31;
    const float* rr = xd + grow * Mz;
    float vals[8], mx = -1e30f;
    #pragma unroll
    for (int i = 0; i < 8; ++i) { vals[i] = rr[lane + i * 32]; mx = fmaxf(mx, vals[i]); }
    #pragma unroll
    for (int o = 16; o; o >>= 1) mx = fmaxf(mx, __shfl_xor(mx, o, 32));
    float dg = diag[grow];
    #pragma unroll
    for (int i = 0; i < 8; ++i)
        qp[grow * Mz + lane + i * 32] = (f16)((__expf(vals[i] - dg - mx) + 1e-3f) * 0.0625f);
}

__global__ __launch_bounds__(256)
void kstab_kernel(const float* __restrict__ xd, float* __restrict__ stab)
{
    __shared__ float red[256];
    long bg = blockIdx.x;
    const float* p = xd + bg * (long)Sz * Mz;
    float mx = -1e30f;
    for (long i = threadIdx.x; i < (long)Sz * Mz; i += 256) mx = fmaxf(mx, p[i]);
    red[threadIdx.x] = mx; __syncthreads();
    for (int s = 128; s; s >>= 1) {
        if ((int)threadIdx.x < s) red[threadIdx.x] = fmaxf(red[threadIdx.x], red[threadIdx.x + s]);
        __syncthreads();
    }
    if (!threadIdx.x) stab[bg] = red[0];
}

__global__ __launch_bounds__(256)
void kfeat_kernel(const float* __restrict__ xd, const float* __restrict__ diag,
                  const float* __restrict__ stab, f16* __restrict__ kp)
{
    long grow = (long)blockIdx.x * 8 + (threadIdx.x >> 5);
    int lane = threadIdx.x & 31;
    float mx = stab[grow / Sz];
    float dg = diag[grow];
    const float* rr = xd + grow * Mz;
    #pragma unroll
    for (int i = 0; i < 8; ++i) {
        int j = lane + i * 32;
        kp[grow * Mz + j] = (f16)((__expf(rr[j] - dg - mx) + 1e-3f) * 0.0625f);
    }
}

// tril mask + intra denominator; sc32 -> sc16
__global__ __launch_bounds__(256)
void sc_mask_den_kernel(const float* __restrict__ sc32, f16* __restrict__ sc16,
                        float* __restrict__ den)
{
    long grow = (long)blockIdx.x * 8 + (threadIdx.x >> 5);   // 0..65535
    int lane = threadIdx.x & 31;
    long bt = grow >> 8; int i = (int)(grow & 255);
    long bg = bt >> 4; int nc = (int)(bt & 15);
    const float* row = sc32 + (bt << 16) + (long)i * 256;
    f16* orow = sc16 + (bt << 16) + (long)i * 256;
    float ssum = 0.f;
    #pragma unroll
    for (int t = 0; t < 8; ++t) {
        int j = lane + t * 32;
        float v = (j <= i) ? row[j] : 0.f;
        orow[j] = (f16)v; ssum += v;
    }
    #pragma unroll
    for (int o = 16; o; o >>= 1) ssum += __shfl_xor(ssum, o, 32);
    if (!lane) den[bg * Sz + (long)nc * CHz + i] = ssum;
}

__global__ void ks_kernel(const f16* __restrict__ kp, float* __restrict__ ks)
{
    int bt = blockIdx.x, m = threadIdx.x;
    int bg = bt >> 4, nc = bt & 15;
    const f16* base = kp + ((long)bg * Sz + (long)nc * CHz) * Mz + m;
    float s = 0.f;
    for (int j = 0; j < CHz; ++j) s += (float)base[(long)j * Mz];
    ks[(long)bt * Mz + m] = s;
}

__global__ void cumsum_kv_kernel(const float* __restrict__ kv, f16* __restrict__ kvp)
{
    long t = (long)blockIdx.x * 256 + threadIdx.x;   // BG*M*64
    int d = (int)(t & 63); int m = (int)((t >> 6) & 255); int bg = (int)(t >> 14);
    float run = 0.f;
    for (int n = 0; n < NCH; ++n) {
        long idx = ((long)(bg * NCH + n) * Mz + m) * 64 + d;
        kvp[idx] = (f16)run; run += kv[idx];
    }
}

__global__ void cumsum_ks_kernel(const float* __restrict__ ks, float* __restrict__ ksp)
{
    long t = (long)blockIdx.x * 256 + threadIdx.x;   // BG*M
    int m = (int)(t & 255); int bg = (int)(t >> 8);
    float run = 0.f;
    for (int n = 0; n < NCH; ++n) {
        long idx = (long)(bg * NCH + n) * Mz + m;
        ksp[idx] = run; run += ks[idx];
    }
}

__global__ __launch_bounds__(256)
void interden_kernel(const f16* __restrict__ qp, const float* __restrict__ ksp,
                     float* __restrict__ den)
{
    long grow = (long)blockIdx.x * 8 + (threadIdx.x >> 5);
    int lane = threadIdx.x & 31;
    long bg = grow / Sz; long s = grow % Sz; int nc = (int)(s >> 8);
    const f16* q = qp + grow * Mz;
    const float* kk = ksp + ((long)bg * NCH + nc) * Mz;
    float a = 0.f;
    #pragma unroll
    for (int i = 0; i < 8; ++i) { int j = lane + i * 32; a += (float)q[j] * kk[j]; }
    #pragma unroll
    for (int o = 16; o; o >>= 1) a += __shfl_xor(a, o, 32);
    if (!lane) den[grow] += a;
}

__global__ void combine_kernel(const float* __restrict__ num, const float* __restrict__ den,
                               f16* __restrict__ outA)
{
    long t = (long)blockIdx.x * 256 + threadIdx.x;   // BG*S*64
    int d = (int)(t & 63); long r = t >> 6;
    long bg = r / Sz, s = r % Sz;
    long b = bg >> 2; int hh = (int)(bg & 3);
    float o = num[r * 64 + d] / den[r];
    outA[((b * Sz + s) * Hz + HLz + hh) * DHz + d] = (f16)o;
}

// GEGLU: gelu(a) * g
__global__ void geglu_kernel(const f16* __restrict__ u, f16* __restrict__ g)
{
    long t = (long)blockIdx.x * 256 + threadIdx.x;   // BS*DFF
    long row = t / DFFz; int c = (int)(t % DFFz);
    float a  = (float)u[row * 2 * DFFz + c];
    float gv = (float)u[row * 2 * DFFz + DFFz + c];
    float gl = 0.5f * a * (1.f + tanhf(0.7978845608f * (a + 0.044715f * a * a * a)));
    g[t] = (f16)(gl * gv);
}

__global__ void zero_kernel(float* __restrict__ o) { o[0] = 0.f; }

// per-row logits + log-softmax + NLL accumulation
__global__ __launch_bounds__(128)
void head_nll_kernel(const float* __restrict__ hf, const float* __restrict__ hw,
                     const float* __restrict__ hb, const int* __restrict__ tgt,
                     float* __restrict__ out)
{
    __shared__ float red[128];
    long row = blockIdx.x; int v = threadIdx.x;
    int cpos = (int)(row & 3);
    const float* hr = hf + row * Dz;
    const float* wc = hw + (long)cpos * Dz * Vz + v;
    float lg = hb[cpos * Vz + v];
    for (int d = 0; d < Dz; ++d) lg += hr[d] * wc[(long)d * Vz];
    red[v] = lg; __syncthreads();
    for (int s = 64; s; s >>= 1) { if (v < s) red[v] = fmaxf(red[v], red[v + s]); __syncthreads(); }
    float mx = red[0]; __syncthreads();
    red[v] = __expf(lg - mx); __syncthreads();
    for (int s = 64; s; s >>= 1) { if (v < s) red[v] += red[v + s]; __syncthreads(); }
    float lse = mx + __logf(red[0]);
    if (v == tgt[row]) atomicAdd(out, (lse - lg) * (1.f / 16384.f));
}

// ============================================================
// Host launcher
// ============================================================
extern "C" void kernel_launch(void* const* d_in, const int* in_sizes, int n_in,
                              void* d_out, int out_size, void* d_ws, size_t ws_size,
                              hipStream_t stream)
{
    (void)in_sizes; (void)n_in; (void)out_size; (void)ws_size;
    // inputs in setup_inputs() order (params dict flattened depth-first)
    const int*   target = (const int*)  d_in[0];
    const float* emb    = (const float*)d_in[1];
    const float* pe     = (const float*)d_in[2];
    const float* linW   = (const float*)d_in[3];
    const float* linb   = (const float*)d_in[4];
    const float* sos    = (const float*)d_in[5];
    const float* ln1g   = (const float*)d_in[6];
    const float* ln1b   = (const float*)d_in[7];
    const float* Wqkv   = (const float*)d_in[8];
    const float* bqkv   = (const float*)d_in[9];
    const float* Wo     = (const float*)d_in[10];
    const float* bo     = (const float*)d_in[11];
    const float* proj   = (const float*)d_in[12];
    const float* ln2g   = (const float*)d_in[13];
    const float* ln2b   = (const float*)d_in[14];
    const float* W1     = (const float*)d_in[15];
    const float* b1     = (const float*)d_in[16];
    const float* W2     = (const float*)d_in[17];
    const float* b2     = (const float*)d_in[18];
    const float* lnfg   = (const float*)d_in[19];
    const float* lnfb   = (const float*)d_in[20];
    const float* headW  = (const float*)d_in[21];
    const float* headb  = (const float*)d_in[22];
    float* out = (float*)d_out;

    // ---- workspace arena ----
    char* cur = (char*)d_ws;
    auto take = [&](size_t bytes) -> char* {
        char* r = cur; cur += (bytes + 255) & ~(size_t)255; return r;
    };
    f16* linW16 = (f16*)take((size_t)128 * Dz * 2);
    f16* Wqkv16 = (f16*)take((size_t)Lz * Dz * 3 * Dz * 2);
    f16* Wo16   = (f16*)take((size_t)Lz * Dz * Dz * 2);
    f16* W116   = (f16*)take((size_t)Lz * Dz * 2 * DFFz * 2);
    f16* W216   = (f16*)take((size_t)Lz * DFFz * Dz * 2);
    f16* proj16 = (f16*)take((size_t)Lz * Mz * DHz * 2);
    f16* A016   = (f16*)take((size_t)BSz * 128 * 2);
    float* lin32 = (float*)take((size_t)BSz * Dz * 4);
    float* x1    = (float*)take((size_t)BSz * Dz * 4);
    float* x2    = (float*)take((size_t)BSz * Dz * 4);
    f16* h16     = (f16*)take((size_t)BSz * Dz * 2);
    f16* qkv16   = (f16*)take((size_t)BSz * 3 * Dz * 2);
    f16* qh16    = (f16*)take((size_t)16 * Sz * 64 * 2);
    f16* kh16    = (f16*)take((size_t)16 * Sz * 64 * 2);
    f16* vh16    = (f16*)take((size_t)16 * Sz * 64 * 2);
    f16* qg16    = (f16*)take((size_t)16 * Sz * 64 * 2);
    f16* kg16    = (f16*)take((size_t)16 * Sz * 64 * 2);
    f16* vg16    = (f16*)take((size_t)16 * Sz * 64 * 2);
    float* diagq = (float*)take((size_t)16 * Sz * 4);
    float* diagk = (float*)take((size_t)16 * Sz * 4);
    float* stabk = (float*)take(256);
    float* big32 = (float*)take((size_t)BGz * Sz * Mz * 4);   // xd32, then sc32
    f16* qp16    = (f16*)take((size_t)BGz * Sz * Mz * 2);
    f16* kp16    = (f16*)take((size_t)BGz * Sz * Mz * 2);
    f16* sc16    = (f16*)take((size_t)BGz * Sz * CHz * 2);
    float* denA  = (float*)take((size_t)BGz * Sz * 4);
    float* intra32 = (float*)take((size_t)BGz * Sz * 64 * 4);
    float* kv32  = (float*)take((size_t)BGz * NCH * Mz * 64 * 4);
    f16* kvp16   = (f16*)take((size_t)BGz * NCH * Mz * 64 * 2);
    float* ks32  = (float*)take((size_t)BGz * NCH * Mz * 4);
    float* ksp32 = (float*)take((size_t)BGz * NCH * Mz * 4);
    f16* attn16  = (f16*)take((size_t)BSz * Dz * 2);
    f16* u16     = (f16*)take((size_t)BSz * 2 * DFFz * 2);
    f16* gg16    = (f16*)take((size_t)BSz * DFFz * 2);
    float* yln   = (float*)take((size_t)BSz * Dz * 4);

    // ---- weight conversions (f32 -> f16) ----
    cvt_kernel<<<512, 256, 0, stream>>>(linW, linW16, (long)128 * Dz);
    cvt_kernel<<<4096, 256, 0, stream>>>(Wqkv, Wqkv16, (long)Lz * Dz * 3 * Dz);
    cvt_kernel<<<4096, 256, 0, stream>>>(Wo,   Wo16,   (long)Lz * Dz * Dz);
    cvt_kernel<<<4096, 256, 0, stream>>>(W1,   W116,   (long)Lz * Dz * 2 * DFFz);
    cvt_kernel<<<4096, 256, 0, stream>>>(W2,   W216,   (long)Lz * DFFz * Dz);
    cvt_kernel<<<512, 256, 0, stream>>>(proj, proj16, (long)Lz * Mz * DHz);

    // ---- embedding -> input linear -> assemble x ----
    embed_kernel<<<(int)(BSz * 128 / 256), 256, 0, stream>>>(target, emb, pe, A016);
    gemm_wmma_kernel<false,false,false,false><<<dim3(BSz / 128, Dz / 64, 1), 256, 0, stream>>>(
        A016, 128, 0, linW16, Dz, 0, linb, lin32, Dz, 0, 128);
    assemble_kernel<<<(int)(BSz * Dz / 256), 256, 0, stream>>>(lin32, sos, x1, x2);

    // ---- layers ----
    for (int l = 0; l < Lz; ++l) {
        const f16* Wqkv_l = Wqkv16 + (size_t)l * Dz * 3 * Dz;
        const f16* Wo_l   = Wo16   + (size_t)l * Dz * Dz;
        const f16* W1_l   = W116   + (size_t)l * Dz * 2 * DFFz;
        const f16* W2_l   = W216   + (size_t)l * DFFz * Dz;
        const f16* proj_l = proj16 + (size_t)l * Mz * DHz;

        // x1 += attn(LN(x2))
        ln_kernel<<<(int)(BSz / 8), 256, 0, stream>>>(x2, ln1g + l * Dz, ln1b + l * Dz, h16);
        gemm_wmma_kernel<false,false,true,false><<<dim3(BSz / 128, 3 * Dz / 64, 1), 256, 0, stream>>>(
            h16, Dz, 0, Wqkv_l, 3 * Dz, 0, bqkv + (long)l * 3 * Dz, qkv16, 3 * Dz, 0, Dz);
        rotary_split_kernel<<<(int)BSz, 512, 0, stream>>>(qkv16, qh16, kh16, vh16,
                                                          qg16, kg16, vg16, diagq, diagk);
        // local attention heads
        local_attn_kernel<<<dim3(16, 16, 16), 256, 0, stream>>>(qh16, kh16, vh16, attn16);

        // performer features: xd = xn * proj^T
        gemm_wmma_kernel<false,true,false,false><<<dim3(BGz * Sz / 128, Mz / 64, 1), 256, 0, stream>>>(
            qg16, DHz, 0, proj_l, DHz, 0, nullptr, big32, Mz, 0, DHz);
        qfeat_kernel<<<(int)(BGz * Sz / 8), 256, 0, stream>>>(big32, diagq, qp16);
        gemm_wmma_kernel<false,true,false,false><<<dim3(BGz * Sz / 128, Mz / 64, 1), 256, 0, stream>>>(
            kg16, DHz, 0, proj_l, DHz, 0, nullptr, big32, Mz, 0, DHz);
        kstab_kernel<<<BGz, 256, 0, stream>>>(big32, stabk);
        kfeat_kernel<<<(int)(BGz * Sz / 8), 256, 0, stream>>>(big32, diagk, stabk, kp16);

        // chunked causal linear attention (batch = BG * NCH = 256)
        gemm_wmma_kernel<false,true,false,false><<<dim3(CHz / 128, CHz / 64, BGz * NCH), 256, 0, stream>>>(
            qp16, Mz, (long)CHz * Mz, kp16, Mz, (long)CHz * Mz, nullptr,
            big32, CHz, (long)CHz * CHz, Mz);                               // sc = qc kc^T
        sc_mask_den_kernel<<<(int)(BGz * Sz / 8), 256, 0, stream>>>(big32, sc16, denA);
        gemm_wmma_kernel<false,false,false,false><<<dim3(CHz / 128, 1, BGz * NCH), 256, 0, stream>>>(
            sc16, CHz, (long)CHz * CHz, vg16, 64, (long)CHz * 64, nullptr,
            intra32, 64, (long)CHz * 64, CHz);                              // intra = sc vc
        gemm_wmma_kernel<true,false,false,false><<<dim3(Mz / 128, 1, BGz * NCH), 256, 0, stream>>>(
            kp16, Mz, (long)CHz * Mz, vg16, 64, (long)CHz * 64, nullptr,
            kv32, 64, (long)Mz * 64, CHz);                                  // kv = kc^T vc
        ks_kernel<<<BGz * NCH, 256, 0, stream>>>(kp16, ks32);
        cumsum_kv_kernel<<<(int)(BGz * Mz * 64 / 256), 256, 0, stream>>>(kv32, kvp16);
        cumsum_ks_kernel<<<(int)(BGz * Mz / 256), 256, 0, stream>>>(ks32, ksp32);
        gemm_wmma_kernel<false,false,false,true><<<dim3(CHz / 128, 1, BGz * NCH), 256, 0, stream>>>(
            qp16, Mz, (long)CHz * Mz, kvp16, 64, (long)Mz * 64, nullptr,
            intra32, 64, (long)CHz * 64, Mz);                               // += inter
        interden_kernel<<<(int)(BGz * Sz / 8), 256, 0, stream>>>(qp16, ksp32, denA);
        combine_kernel<<<(int)(BGz * Sz * 64 / 256), 256, 0, stream>>>(intra32, denA, attn16);

        // output projection, accumulate into residual x1
        gemm_wmma_kernel<false,false,false,true><<<dim3(BSz / 128, Dz / 64, 1), 256, 0, stream>>>(
            attn16, Dz, 0, Wo_l, Dz, 0, bo + (long)l * Dz, x1, Dz, 0, Dz);

        // x2 += ff(LN(x1))
        ln_kernel<<<(int)(BSz / 8), 256, 0, stream>>>(x1, ln2g + l * Dz, ln2b + l * Dz, h16);
        gemm_wmma_kernel<false,false,true,false><<<dim3(BSz / 128, 2 * DFFz / 64, 1), 256, 0, stream>>>(
            h16, Dz, 0, W1_l, 2 * DFFz, 0, b1 + (long)l * 2 * DFFz, u16, 2 * DFFz, 0, Dz);
        geglu_kernel<<<(int)(BSz * DFFz / 256), 256, 0, stream>>>(u16, gg16);
        gemm_wmma_kernel<false,false,false,true><<<dim3(BSz / 128, Dz / 64, 1), 256, 0, stream>>>(
            gg16, DFFz, 0, W2_l, Dz, 0, b2 + (long)l * Dz, x2, Dz, 0, DFFz);
    }

    // ---- final LN + head NLL ----
    final_ln_kernel<<<(int)(BSz / 8), 256, 0, stream>>>(x1, x2, lnfg, lnfb, yln);
    zero_kernel<<<1, 1, 0, stream>>>(out);
    head_nll_kernel<<<(int)BSz, 128, 0, stream>>>(yln, headW, headb, target, out);
}
